// HGNN_2508260901597
// MI455X (gfx1250) — compile-verified, matrix operand order checked by reading, not run on hardware
//
#include <hip/hip_runtime.h>
#include <hip/hip_bf16.h>
#include <math.h>

#define NN      131072
#define DEG     16
#define NNZ     (NN * DEG)
#define DMODEL  64
#define NHEAD   4
#define HDIM    16
#define BLK     256

typedef __attribute__((ext_vector_type(16))) _Float16 v16h;
typedef __attribute__((ext_vector_type(8)))  _Float16 v8h;
typedef __attribute__((ext_vector_type(8)))  float    v8f;

// CDNA5 16-bit A/B fragment K-mapping: half h (0..15), hi = (lane>=16)
// K = (h&7) + 8*((h>>3)*2 + hi)  -> per lane: two contiguous runs of 8
__device__ __forceinline__ v8f v8f_zero() {
    v8f z = {0.f, 0.f, 0.f, 0.f, 0.f, 0.f, 0.f, 0.f};
    return z;
}

__device__ __forceinline__ v16h cat16(v8h a, v8h b) {
    return __builtin_shufflevector(a, b, 0, 1, 2, 3, 4, 5, 6, 7,
                                   8, 9, 10, 11, 12, 13, 14, 15);
}

__device__ __forceinline__ v16h frag_from_f32(float4 a0, float4 a1,
                                              float4 a2, float4 a3) {
    v16h r;
    r[0] = (_Float16)a0.x;  r[1] = (_Float16)a0.y;
    r[2] = (_Float16)a0.z;  r[3] = (_Float16)a0.w;
    r[4] = (_Float16)a1.x;  r[5] = (_Float16)a1.y;
    r[6] = (_Float16)a1.z;  r[7] = (_Float16)a1.w;
    r[8] = (_Float16)a2.x;  r[9] = (_Float16)a2.y;
    r[10] = (_Float16)a2.z; r[11] = (_Float16)a2.w;
    r[12] = (_Float16)a3.x; r[13] = (_Float16)a3.y;
    r[14] = (_Float16)a3.z; r[15] = (_Float16)a3.w;
    return r;
}

__device__ __forceinline__ float4 relu4(float4 a, float4 b) {
    float4 r;
    r.x = a.x + b.x; r.x = r.x > 0.f ? r.x : 0.f;
    r.y = a.y + b.y; r.y = r.y > 0.f ? r.y : 0.f;
    r.z = a.z + b.z; r.z = r.z > 0.f ? r.z : 0.f;
    r.w = a.w + b.w; r.w = r.w > 0.f ? r.w : 0.f;
    return r;
}

// ---------------------------------------------------------------------------
// WMMA GEMM: C[M,Nc] = act_out( act_in(A + abias) @ W + obias )
// A row-major [M,K] f32; W row-major [K,Nc] f32, staged f16-transposed in LDS.
// One wave computes TM sequential 16-row tiles x (16*TN) columns; A fragment
// reused across TN WMMAs; the LDS W panel amortized over TM row tiles.
// Requires K%32==0, Nc%(16*TN)==0, M%(16*TM)==0, K*Nc <= 12288.
// ---------------------------------------------------------------------------
template <int TM, int TN, bool RELU_IN, int ACT_OUT>
__global__ void wmma_gemm_kernel(const float* __restrict__ A,
                                 const float* __restrict__ abias,
                                 const float* __restrict__ W,
                                 const float* __restrict__ obias,
                                 float* __restrict__ C,
                                 int M, int K, int Nc)
{
    __shared__ __align__(16) _Float16 sW[12288];   // [n][k] transposed, 24 KB max

    int tid  = (int)threadIdx.x;
    int lane = tid & 31;
    int hi = lane >> 4, lo = lane & 15;

    // cooperative load of W, transposed + converted to f16 (coalesced reads)
    for (int i = tid; i < K * Nc; i += 256) {
        int k = i / Nc, n = i - k * Nc;
        sW[n * K + k] = (_Float16)W[i];
    }
    __syncthreads();

    int gwid = (int)((blockIdx.x * blockDim.x + threadIdx.x) >> 5);
    int groups_n = Nc / (16 * TN);
    int total = (M / (16 * TM)) * groups_n;
    if (gwid >= total) return;                      // wave-uniform
    int mg = gwid / groups_n;
    int tg = gwid - mg * groups_n;

    #pragma unroll
    for (int t = 0; t < TM; ++t) {
        int tm = mg * TM + t;
        int m = tm * 16 + lo;

        v8f c[TN];
        #pragma unroll
        for (int j = 0; j < TN; ++j) c[j] = v8f_zero();

        const float* Arow = A + (size_t)m * K;
        for (int k0 = 0; k0 < K; k0 += 32) {
            // A fragment: two contiguous runs of 8 floats
            const float4* ap0 = (const float4*)(Arow + k0 + 8 * hi);
            const float4* ap1 = (const float4*)(Arow + k0 + 16 + 8 * hi);
            float4 a0 = ap0[0], a1 = ap0[1], a2 = ap1[0], a3 = ap1[1];
            if (RELU_IN) {
                const float4* bp0 = (const float4*)(abias + k0 + 8 * hi);
                const float4* bp1 = (const float4*)(abias + k0 + 16 + 8 * hi);
                a0 = relu4(a0, bp0[0]); a1 = relu4(a1, bp0[1]);
                a2 = relu4(a2, bp1[0]); a3 = relu4(a3, bp1[1]);
            }
            v16h af = frag_from_f32(a0, a1, a2, a3);

            #pragma unroll
            for (int j = 0; j < TN; ++j) {
                int n = (tg * TN + j) * 16 + lo;
                const v8h* wp0 = (const v8h*)(&sW[n * K + k0 + 8 * hi]);
                const v8h* wp1 = (const v8h*)(&sW[n * K + k0 + 16 + 8 * hi]);
                v16h bf = cat16(wp0[0], wp1[0]);
                c[j] = __builtin_amdgcn_wmma_f32_16x16x32_f16(
                    false, af, false, bf, (short)0, c[j], false, false);
            }
        }

        #pragma unroll
        for (int j = 0; j < TN; ++j) {
            int n = (tg * TN + j) * 16 + lo;
            #pragma unroll
            for (int r = 0; r < 8; ++r) {
                int row = tm * 16 + r + 8 * hi;
                float v = c[j][r];
                if (obias) v += obias[n];
                if (ACT_OUT == 1) {
                    v = v > 0.f ? v : 0.f;
                } else if (ACT_OUT == 2) {
                    v = 0.5f * v * (1.0f + erff(v * 0.70710678118654752f));
                }
                C[(size_t)row * Nc + n] = v;
            }
        }
    }
}

// ---------------------------------------------------------------------------
// SpMM scatter: out[rows[e], :] += vals[e] * in[cols[e], :]
// ---------------------------------------------------------------------------
__global__ void spmm_kernel(const int* __restrict__ rows,
                            const int* __restrict__ cols,
                            const float* __restrict__ vals,
                            const float* __restrict__ in,
                            float* __restrict__ out, int F4)
{
    long long t = (long long)blockIdx.x * blockDim.x + threadIdx.x;
    long long total = (long long)NNZ * F4;
    if (t >= total) return;
    int e  = (int)(t / F4);
    int c4 = (int)(t - (long long)e * F4);
    int F  = F4 * 4;
    float v = vals[e];
    const float4 x = ((const float4*)(in + (size_t)cols[e] * F))[c4];
    float* o = out + (size_t)rows[e] * F + c4 * 4;
    atomicAdd(o + 0, v * x.x);
    atomicAdd(o + 1, v * x.y);
    atomicAdd(o + 2, v * x.z);
    atomicAdd(o + 3, v * x.w);
}

__global__ void zero_kernel(float* __restrict__ p, int n)
{
    int i = (int)(blockIdx.x * blockDim.x + threadIdx.x);
    if (i < n) p[i] = 0.f;
}

__global__ void bias_relu_kernel(const float* __restrict__ in,
                                 const float* __restrict__ bias,
                                 float* __restrict__ out, int total, int F)
{
    int i = (int)(blockIdx.x * blockDim.x + threadIdx.x);
    if (i < total) {
        float v = in[i] + bias[i % F];
        out[i] = v > 0.f ? v : 0.f;
    }
}

// ---------------------------------------------------------------------------
// Block self-attention, online-softmax (flash) variant.
// One workgroup per (block b, head h). Live registers per wave: Q frag (8),
// ctx (8), 2 score tiles (16), running m/l (16) -> no spills.
// head_dim=16 zero-padded to WMMA K=32 (halves 8..15 are zero).
// ---------------------------------------------------------------------------
__global__ void attn_kernel(const float* __restrict__ QKV,
                            float* __restrict__ ATT)
{
    __shared__ __align__(16) _Float16 sK[BLK * HDIM];   // [key][d]  8 KB
    __shared__ __align__(16) _Float16 sVt[HDIM * BLK];  // [d][key]  8 KB
    __shared__ __align__(16) _Float16 sP[8][16 * 32];   // per-wave P staging

    int b = (int)(blockIdx.x >> 2);
    int h = (int)(blockIdx.x & 3);
    int tid  = (int)threadIdx.x;
    int wid  = tid >> 5;
    int lane = tid & 31;
    int hi   = lane >> 4;
    int lo   = lane & 15;
    size_t base = (size_t)b * BLK * (3 * DMODEL);

    for (int i = tid; i < BLK * HDIM; i += 256) {
        int s = i >> 4, d = i & 15;
        sK[i]            = (_Float16)QKV[base + (size_t)s * 192 + DMODEL     + h * HDIM + d];
        sVt[d * BLK + s] = (_Float16)QKV[base + (size_t)s * 192 + 2 * DMODEL + h * HDIM + d];
    }
    __syncthreads();

    v8h zero8 = {(_Float16)0.f, (_Float16)0.f, (_Float16)0.f, (_Float16)0.f,
                 (_Float16)0.f, (_Float16)0.f, (_Float16)0.f, (_Float16)0.f};

    for (int qt = wid; qt < 16; qt += 8) {
        int qrow = qt * 16 + lo;
        // Q fragment, pre-scaled by 1/sqrt(16); upper 8 halves (k>=16) zero
        const float4* qp = (const float4*)(QKV + base + (size_t)qrow * 192
                                           + h * HDIM + 8 * hi);
        float4 q0 = qp[0], q1 = qp[1];
        q0.x *= 0.25f; q0.y *= 0.25f; q0.z *= 0.25f; q0.w *= 0.25f;
        q1.x *= 0.25f; q1.y *= 0.25f; q1.z *= 0.25f; q1.w *= 0.25f;
        float4 zf = {0.f, 0.f, 0.f, 0.f};
        v16h qf = frag_from_f32(q0, q1, zf, zf);

        v8f ctx = v8f_zero();
        float m[8], l[8];
        #pragma unroll
        for (int r = 0; r < 8; ++r) { m[r] = -3.0e38f; l[r] = 0.f; }

        for (int kk = 0; kk < 8; ++kk) {
            // two 16-wide score tiles (keys kk*32 .. kk*32+31)
            v8f s0, s1;
            {
                int key = (kk * 2) * 16 + lo;
                v16h bf = cat16(*(const v8h*)(&sK[key * HDIM + 8 * hi]), zero8);
                v8f z = v8f_zero();
                s0 = __builtin_amdgcn_wmma_f32_16x16x32_f16(
                    false, qf, false, bf, (short)0, z, false, false);
            }
            {
                int key = (kk * 2 + 1) * 16 + lo;
                v16h bf = cat16(*(const v8h*)(&sK[key * HDIM + 8 * hi]), zero8);
                v8f z = v8f_zero();
                s1 = __builtin_amdgcn_wmma_f32_16x16x32_f16(
                    false, qf, false, bf, (short)0, z, false, false);
            }
            // online softmax update per row r+8*hi (16-lane row groups).
            // args of __expf are <= 0 after max subtraction: fast path is safe.
            #pragma unroll
            for (int r = 0; r < 8; ++r) {
                float tm = fmaxf(s0[r], s1[r]);
                #pragma unroll
                for (int d = 1; d < 16; d <<= 1)
                    tm = fmaxf(tm, __shfl_xor(tm, d, 32));
                float nm = fmaxf(m[r], tm);
                float sc = __expf(m[r] - nm);
                m[r] = nm;
                float e0 = __expf(s0[r] - nm);
                float e1 = __expf(s1[r] - nm);
                float rs = e0 + e1;
                #pragma unroll
                for (int d = 1; d < 16; d <<= 1)
                    rs += __shfl_xor(rs, d, 32);
                l[r] = l[r] * sc + rs;
                ctx[r] *= sc;
                int row = r + 8 * hi;
                sP[wid][row * 32 + lo]      = (_Float16)e0;
                sP[wid][row * 32 + 16 + lo] = (_Float16)e1;
            }
            // ctx += P @ V (per-wave LDS staging; same-wave DS ordering)
            v16h af = cat16(*(const v8h*)(&sP[wid][lo * 32 + 8 * hi]),
                            *(const v8h*)(&sP[wid][lo * 32 + 16 + 8 * hi]));
            v16h bf = cat16(*(const v8h*)(&sVt[lo * BLK + kk * 32 + 8 * hi]),
                            *(const v8h*)(&sVt[lo * BLK + kk * 32 + 16 + 8 * hi]));
            ctx = __builtin_amdgcn_wmma_f32_16x16x32_f16(
                false, af, false, bf, (short)0, ctx, false, false);
        }

        #pragma unroll
        for (int r = 0; r < 8; ++r) {
            int row = b * BLK + qt * 16 + r + 8 * hi;
            ATT[(size_t)row * DMODEL + h * HDIM + lo] = ctx[r] / l[r];
        }
    }
}

// ---------------------------------------------------------------------------
// Residual + LayerNorm over D=64; one wave per row (wave32: 2 elems/lane).
// ---------------------------------------------------------------------------
__global__ void ln_residual_kernel(const float* __restrict__ X,
                                   const float* __restrict__ Y,
                                   const float* __restrict__ g,
                                   const float* __restrict__ bt,
                                   float* __restrict__ out, int Mrows)
{
    int w = (int)((blockIdx.x * blockDim.x + threadIdx.x) >> 5);
    int lane = (int)(threadIdx.x & 31);
    if (w >= Mrows) return;
    size_t off = (size_t)w * DMODEL;
    float v0 = X[off + lane]      + Y[off + lane];
    float v1 = X[off + lane + 32] + Y[off + lane + 32];
    float s = v0 + v1;
    #pragma unroll
    for (int d = 1; d < 32; d <<= 1) s += __shfl_xor(s, d, 32);
    float mu = s * (1.f / 64.f);
    float d0 = v0 - mu, d1 = v1 - mu;
    float vs = d0 * d0 + d1 * d1;
    #pragma unroll
    for (int d = 1; d < 32; d <<= 1) vs += __shfl_xor(vs, d, 32);
    float rstd = rsqrtf(vs * (1.f / 64.f) + 1e-5f);
    out[off + lane]      = d0 * rstd * g[lane]      + bt[lane];
    out[off + lane + 32] = d1 * rstd * g[lane + 32] + bt[lane + 32];
}

// ---------------------------------------------------------------------------
// Classifier + log_softmax (fp32). One wave per row.
// ---------------------------------------------------------------------------
__global__ void cls_kernel(const float* __restrict__ X2,
                           const float* __restrict__ Wc,
                           const float* __restrict__ bc,
                           float* __restrict__ out, int Mrows)
{
    int w = (int)((blockIdx.x * blockDim.x + threadIdx.x) >> 5);
    int lane = (int)(threadIdx.x & 31);
    int hi = lane >> 4, lo = lane & 15;
    if (w >= Mrows) return;
    size_t off = (size_t)w * DMODEL;
    float acc = 0.f;
    #pragma unroll
    for (int kk = 0; kk < 32; ++kk) {
        int k = hi * 32 + kk;
        acc += X2[off + k] * Wc[k * 16 + lo];
    }
    acc += __shfl_xor(acc, 16, 32);
    acc += bc[lo];
    float mx = acc;
    #pragma unroll
    for (int d = 1; d < 16; d <<= 1) mx = fmaxf(mx, __shfl_xor(mx, d, 32));
    float e = expf(acc - mx), sm = e;
    #pragma unroll
    for (int d = 1; d < 16; d <<= 1) sm += __shfl_xor(sm, d, 32);
    float res = acc - mx - logf(sm);
    if (hi == 0) out[(size_t)w * 16 + lo] = res;
}

// ---------------------------------------------------------------------------

static inline int ceil_div(long long a, long long b) { return (int)((a + b - 1) / b); }

#define GEMM_WAVES(M, TM, Nc, TN) \
    ((long long)((M) / (16 * (TM))) * ((Nc) / (16 * (TN))))

static void launch_gemm_none(const float* A, const float* W, const float* ob,
                             float* C, int M, int K, int Nc, hipStream_t s)
{
    long long waves = GEMM_WAVES(M, 4, Nc, 4);
    wmma_gemm_kernel<4, 4, false, 0><<<ceil_div(waves * 32, 256), 256, 0, s>>>(
        A, nullptr, W, ob, C, M, K, Nc);
}

extern "C" void kernel_launch(void* const* d_in, const int* in_sizes, int n_in,
                              void* d_out, int out_size, void* d_ws, size_t ws_size,
                              hipStream_t stream)
{
    const float* x      = (const float*)d_in[0];
    const int*   G_rows = (const int*)  d_in[1];
    const int*   G_cols = (const int*)  d_in[2];
    const float* G_vals = (const float*)d_in[3];
    const float* W1     = (const float*)d_in[4];
    const float* b1     = (const float*)d_in[5];
    const float* W2     = (const float*)d_in[6];
    const float* b2     = (const float*)d_in[7];
    const float* W_qkv  = (const float*)d_in[8];
    const float* b_qkv  = (const float*)d_in[9];
    const float* W_o    = (const float*)d_in[10];
    const float* b_o    = (const float*)d_in[11];
    const float* ln1_g  = (const float*)d_in[12];
    const float* ln1_b  = (const float*)d_in[13];
    const float* W_ff1  = (const float*)d_in[14];
    const float* b_ff1  = (const float*)d_in[15];
    const float* W_ff2  = (const float*)d_in[16];
    const float* b_ff2  = (const float*)d_in[17];
    const float* ln2_g  = (const float*)d_in[18];
    const float* ln2_b  = (const float*)d_in[19];
    const float* W_cls  = (const float*)d_in[20];
    const float* b_cls  = (const float*)d_in[21];
    float* out = (float*)d_out;

    float* ws = (float*)d_ws;
    const size_t n = NN;
    // workspace layout with reuse; peak = N*448 floats = 224 MB
    float* P1  = ws;                 // [N,32]
    float* S1  = ws + n * 32;        // [N,32]
    float* P2  = ws + n * 64;        // [N,64]
    float* S2  = ws + n * 128;       // [N,64]
    float* H2  = ws + n * 192;       // [N,64]   (live until LN1)
    float* QKV = ws + n * 256;       // [N,192]
    float* ATT = ws;                 // [N,64]   reuse P1/S1
    float* A1  = ws + n * 64;        // [N,64]   reuse P2
    float* X1  = ws + n * 128;       // [N,64]   reuse S2
    float* FF1 = ws + n * 256;       // [N,128]  reuse QKV
    float* A2  = ws + n * 384;       // [N,64]
    float* X2  = ws;                 // [N,64]   reuse ATT

    // 1) P1 = X @ W1   (projection commuted ahead of the gather: 4x less SpMM traffic)
    {
        long long waves = GEMM_WAVES(NN, 4, 32, 2);
        wmma_gemm_kernel<4, 2, false, 0><<<ceil_div(waves * 32, 256), 256, 0, stream>>>(
            x, nullptr, W1, nullptr, P1, NN, 128, 32);
    }

    // 2) S1 = G @ P1  (zero + atomic scatter, 32 channels)
    zero_kernel<<<ceil_div((long long)NN * 32, 256), 256, 0, stream>>>(S1, NN * 32);
    spmm_kernel<<<ceil_div((long long)NNZ * 8, 256), 256, 0, stream>>>(
        G_rows, G_cols, G_vals, P1, S1, 8);

    // 3) P2 = relu(S1 + b1) @ W2   (fused input bias+relu)
    {
        long long waves = GEMM_WAVES(NN, 4, 64, 4);
        wmma_gemm_kernel<4, 4, true, 0><<<ceil_div(waves * 32, 256), 256, 0, stream>>>(
            S1, b1, W2, nullptr, P2, NN, 32, 64);
    }

    // 4) S2 = G @ P2  (64 channels)
    zero_kernel<<<ceil_div((long long)NN * 64, 256), 256, 0, stream>>>(S2, NN * 64);
    spmm_kernel<<<ceil_div((long long)NNZ * 16, 256), 256, 0, stream>>>(
        G_rows, G_cols, G_vals, P2, S2, 16);

    // 5) H2 = relu(S2 + b2)  (materialized: needed for residual)
    bias_relu_kernel<<<ceil_div((long long)NN * 64, 256), 256, 0, stream>>>(
        S2, b2, H2, NN * 64, 64);

    // 6) QKV = H2 @ W_qkv + b_qkv
    launch_gemm_none(H2, W_qkv, b_qkv, QKV, NN, 64, 192, stream);

    // 7) block attention -> ATT
    attn_kernel<<<(NN / BLK) * NHEAD, 256, 0, stream>>>(QKV, ATT);

    // 8) A1 = ATT @ W_o + b_o
    launch_gemm_none(ATT, W_o, b_o, A1, NN, 64, 64, stream);

    // 9) X1 = LN1(H2 + A1)
    ln_residual_kernel<<<ceil_div((long long)NN * 32, 256), 256, 0, stream>>>(
        H2, A1, ln1_g, ln1_b, X1, NN);

    // 10) FF1 = gelu(X1 @ W_ff1 + b_ff1)
    {
        long long waves = GEMM_WAVES(NN, 4, 128, 4);
        wmma_gemm_kernel<4, 4, false, 2><<<ceil_div(waves * 32, 256), 256, 0, stream>>>(
            X1, nullptr, W_ff1, b_ff1, FF1, NN, 64, 128);
    }

    // 11) A2 = FF1 @ W_ff2 + b_ff2
    launch_gemm_none(FF1, W_ff2, b_ff2, A2, NN, 128, 64, stream);

    // 12) X2 = LN2(X1 + A2)
    ln_residual_kernel<<<ceil_div((long long)NN * 32, 256), 256, 0, stream>>>(
        X1, A2, ln2_g, ln2_b, X2, NN);

    // 13) log_softmax(X2 @ W_cls + b_cls) -> out
    cls_kernel<<<ceil_div((long long)NN * 32, 256), 256, 0, stream>>>(
        X2, W_cls, b_cls, out, NN);
}
